// MLA_74560632259154
// MI455X (gfx1250) — matrix operand order repaired and checked by reading, not compiled, not run on hardware
//
#include <hip/hip_runtime.h>

typedef unsigned short u16;
typedef __attribute__((ext_vector_type(8)))  __bf16 v8bf;
typedef __attribute__((ext_vector_type(16))) __bf16 v16bf;
typedef __attribute__((ext_vector_type(8)))  float  v8f;
typedef __attribute__((ext_vector_type(16))) unsigned short u16x16;

#define BSZ   2
#define SEQ   2048
#define DIM   2048
#define DDOWN 512
#define DUP   1024
#define NH    16
#define RDIM  32
#define CDIM  64
#define VDIM  64
#define NROW  (BSZ*SEQ)      // 4096
#define QKD   (CDIM + RDIM)  // 96
#define SCALE_F 16.970562748477143f   // sqrt(128)+sqrt(32)

// ---------- helpers ----------
static __device__ __forceinline__ u16 f2bf(float f) {
  unsigned u = __builtin_bit_cast(unsigned, f);
  u += 0x7FFFu + ((u >> 16) & 1u);       // round-to-nearest-even
  return (u16)(u >> 16);
}
static __device__ __forceinline__ float bf2f(u16 h) {
  unsigned u = ((unsigned)h) << 16;
  return __builtin_bit_cast(float, u);
}
static __device__ __forceinline__ v16bf cat88(v8bf a, v8bf b) {
  return __builtin_shufflevector(a, b, 0,1,2,3,4,5,6,7,8,9,10,11,12,13,14,15);
}
static __device__ __forceinline__ v8f wmma_bf16(v16bf a, v16bf b, v8f c) {
  return __builtin_amdgcn_wmma_f32_16x16x32_bf16(false, a, false, b, (short)0, c, false, false);
}
// Async copy of 32 contiguous bf16 (64B) global -> LDS, per lane.
// INST_OFFSET is added to BOTH the LDS and global address (ISA 08_async_tensor §4.4).
static __device__ __forceinline__ void async_copy64B(unsigned lds_addr, const void* gsrc) {
  asm volatile(
      "global_load_async_to_lds_b128 %0, %1, off\n\t"
      "global_load_async_to_lds_b128 %0, %1, off offset:16\n\t"
      "global_load_async_to_lds_b128 %0, %1, off offset:32\n\t"
      "global_load_async_to_lds_b128 %0, %1, off offset:48"
      :: "v"(lds_addr), "v"(gsrc) : "memory");
}
static __device__ __forceinline__ void async_copy32B(unsigned lds_addr, const void* gsrc) {
  asm volatile(
      "global_load_async_to_lds_b128 %0, %1, off\n\t"
      "global_load_async_to_lds_b128 %0, %1, off offset:16"
      :: "v"(lds_addr), "v"(gsrc) : "memory");
}
static __device__ __forceinline__ void wait_async0() {
  asm volatile("s_wait_asynccnt 0x0" ::: "memory");
}
static __device__ __forceinline__ void wait_ds0() {
  asm volatile("s_wait_dscnt 0x0" ::: "memory");
}
static __device__ __forceinline__ unsigned lds_addr_of(const void* p) {
  return (unsigned)(unsigned long long)p;   // LDS flat addr truncates to LDS offset
}
// A-fragment (16x32 bf16, M x K) from row-major [m][k] storage.
// lane<16: K {0..15}; lane>=16: K {16..31}; m = lane&15.
static __device__ __forceinline__ v16bf frag_a_rowmajor(const u16* base, int stride) {
  const int lane = threadIdx.x & 31;
  const u16* row = base + (size_t)(lane & 15) * stride;
  const int hi = (lane >> 4) * 16;
  v8bf lo = *(const v8bf*)(row + hi);
  v8bf hh = *(const v8bf*)(row + 8 + hi);
  return cat88(lo, hh);
}
// B-fragment (32x16 bf16, K x N) where source is stored [n][k] row-major
// (i.e. weights W[N,K], or K-cache rows): b[i] = B[n][kbase+i], contiguous.
static __device__ __forceinline__ v16bf frag_b_rowk(const u16* base, int stride) {
  const int lane = threadIdx.x & 31;
  const u16* row = base + (size_t)(lane & 15) * stride + (lane >> 4) * 16;
  v8bf lo = *(const v8bf*)(row);
  v8bf hh = *(const v8bf*)(row + 8);
  return cat88(lo, hh);
}

// ---------- elementwise kernels ----------
__global__ void cvt_f32_bf16_kernel(const float* __restrict__ s, u16* __restrict__ d, int n) {
  int i = blockIdx.x * 256 + threadIdx.x;
  if (i < n) d[i] = f2bf(s[i]);
}

// src: [NROW, NH*C] row-major -> dst: [B, NH, SEQ, dstride] (writes cols [0,C))
__global__ void scatter_heads_kernel(const u16* __restrict__ src, u16* __restrict__ dst,
                                     int C, int dstride) {
  int idx = blockIdx.x * 256 + threadIdx.x;
  if (idx >= NROW * NH * C) return;
  int c  = idx % C;
  int h  = (idx / C) % NH;
  int bs = idx / (C * NH);
  int b = bs / SEQ, s = bs % SEQ;
  dst[((size_t)((b * NH + h) * SEQ + s)) * dstride + c] = src[idx];
}

// qr_flat [NROW, NH*RDIM] -> rope -> Q[b,h,s, 64..95]
__global__ void rope_q_kernel(const u16* __restrict__ qr, u16* __restrict__ Qb) {
  int idx = blockIdx.x * 256 + threadIdx.x;
  if (idx >= NROW * NH * 16) return;
  int i  = idx & 15;
  int h  = (idx >> 4) & (NH - 1);
  int bs = idx >> 8;                  // /(16*NH)
  int b = bs / SEQ, s = bs % SEQ;
  float inv = __powf(10000.f, -(float)i * (1.f / 16.f));
  float ang = (float)s * inv;
  float sn, cs; __sincosf(ang, &sn, &cs);
  const u16* row = qr + (size_t)bs * (NH * RDIM) + h * RDIM;
  float x1 = bf2f(row[i]), x2 = bf2f(row[16 + i]);
  u16* o = Qb + ((size_t)((b * NH + h) * SEQ + s)) * QKD + CDIM;
  o[i]      = f2bf(x1 * cs - x2 * sn);
  o[16 + i] = f2bf(x2 * cs + x1 * sn);
}

// kr_flat [NROW, RDIM] -> rope -> broadcast over heads into K[b,h,s, 64..95]
__global__ void rope_k_bcast_kernel(const u16* __restrict__ kr, u16* __restrict__ Kb) {
  int idx = blockIdx.x * 256 + threadIdx.x;
  if (idx >= NROW * 16) return;
  int i  = idx & 15;
  int bs = idx >> 4;
  int b = bs / SEQ, s = bs % SEQ;
  float inv = __powf(10000.f, -(float)i * (1.f / 16.f));
  float ang = (float)s * inv;
  float sn, cs; __sincosf(ang, &sn, &cs);
  const u16* row = kr + (size_t)bs * RDIM;
  float x1 = bf2f(row[i]), x2 = bf2f(row[16 + i]);
  u16 y1 = f2bf(x1 * cs - x2 * sn);
  u16 y2 = f2bf(x2 * cs + x1 * sn);
  #pragma unroll
  for (int h = 0; h < NH; ++h) {
    u16* o = Kb + ((size_t)((b * NH + h) * SEQ + s)) * QKD + CDIM;
    o[i] = y1; o[16 + i] = y2;
  }
}

// ---------- tiled WMMA GEMM: C[M,N] = A[M,K] @ B[N,K]^T + bias ----------
// block = 256 threads = 8 waves (4 M x 2 N), block tile 128x128, K step 32.
// Tile staging uses CDNA5 async global->LDS DMA (ASYNCcnt).
__global__ __launch_bounds__(256) void gemm_bf16_kernel(
    const u16* __restrict__ A, const u16* __restrict__ B, const float* __restrict__ bias,
    u16* __restrict__ Cbf, float* __restrict__ Cf, int M, int N, int K)
{
  __shared__ __align__(16) u16 Al[128 * 32];
  __shared__ __align__(16) u16 Bl[128 * 32];
  const int tid  = threadIdx.x;
  const int lane = tid & 31;
  const int wv   = tid >> 5;
  const int wm   = wv >> 1, wn = wv & 1;
  const int m0 = blockIdx.y * 128, n0 = blockIdx.x * 128;
  const v8f z = {0.f,0.f,0.f,0.f,0.f,0.f,0.f,0.f};
  v8f acc[2][4];
  #pragma unroll
  for (int a = 0; a < 2; ++a)
    #pragma unroll
    for (int b = 0; b < 4; ++b) acc[a][b] = z;

  const int srow = tid >> 1;
  const int sseg = (tid & 1) * 16;
  const int gnB  = n0 + srow;
  const unsigned ldsA = lds_addr_of(&Al[srow * 32 + sseg]);
  const unsigned ldsB = lds_addr_of(&Bl[srow * 32 + sseg]);

  for (int k0 = 0; k0 < K; k0 += 32) {
    // async-stage A tile (M % 128 == 0: always in range)
    async_copy32B(ldsA, A + (size_t)(m0 + srow) * K + k0 + sseg);
    // async-stage B tile (guard N; zero-fill out-of-range rows)
    if (gnB < N) {
      async_copy32B(ldsB, B + (size_t)gnB * K + k0 + sseg);
    } else {
      float4 zz = make_float4(0.f, 0.f, 0.f, 0.f);
      *(float4*)(&Bl[srow * 32 + sseg])     = zz;
      *(float4*)(&Bl[srow * 32 + sseg + 8]) = zz;
    }
    wait_async0();
    __syncthreads();
    v16bf af0 = frag_a_rowmajor(&Al[(wm * 32 + 0 ) * 32], 32);
    v16bf af1 = frag_a_rowmajor(&Al[(wm * 32 + 16) * 32], 32);
    #pragma unroll
    for (int nt = 0; nt < 4; ++nt) {
      v16bf bf = frag_b_rowk(&Bl[(wn * 64 + nt * 16) * 32], 32);
      acc[0][nt] = wmma_bf16(af0, bf, acc[0][nt]);
      acc[1][nt] = wmma_bf16(af1, bf, acc[1][nt]);
    }
    __syncthreads();
  }
  const int col = lane & 15;
  const int rb  = (lane >> 4) * 8;
  #pragma unroll
  for (int mt = 0; mt < 2; ++mt)
    #pragma unroll
    for (int nt = 0; nt < 4; ++nt) {
      int gn = n0 + wn * 64 + nt * 16 + col;
      if (gn >= N) continue;
      float bv = bias[gn];
      #pragma unroll
      for (int j = 0; j < 8; ++j) {
        int gm = m0 + wm * 32 + mt * 16 + rb + j;
        float v = acc[mt][nt][j] + bv;
        if (Cf) Cf[(size_t)gm * N + gn] = v;
        else    Cbf[(size_t)gm * N + gn] = f2bf(v);
      }
    }
}

// ---------- flash attention: one wave per 16-query tile ----------
__global__ __launch_bounds__(256) void mla_attn_kernel(
    const u16* __restrict__ Qg, const u16* __restrict__ Kg, const u16* __restrict__ Vg,
    u16* __restrict__ Og)
{
  __shared__ __align__(16) u16 Vl[8][32 * VDIM];  // per-wave V block (32 keys x 64)
  __shared__ __align__(16) u16 Pl[8][16 * 32];    // per-wave P tile
  const int lane = threadIdx.x & 31;
  const int wv   = threadIdx.x >> 5;
  const int tile = blockIdx.x * 8 + wv;
  const int ntq  = SEQ / 16;
  const int q0   = (tile % ntq) * 16;
  const int bh   = tile / ntq;
  const int b = bh / NH, hh = bh % NH;
  const u16* Q  = Qg + (size_t)bh * SEQ * QKD;
  const u16* Kc = Kg + (size_t)bh * SEQ * QKD;
  const u16* Vc = Vg + (size_t)bh * SEQ * VDIM;

  // Q A-fragments for the 3 k-steps of d = 96
  v16bf qf[3];
  {
    const u16* row = Q + (size_t)(q0 + (lane & 15)) * QKD;
    const int hi = (lane >> 4) * 16;
    #pragma unroll
    for (int ks = 0; ks < 3; ++ks)
      qf[ks] = cat88(*(const v8bf*)(row + ks * 32 + hi),
                     *(const v8bf*)(row + ks * 32 + 8 + hi));
  }

  const v8f z = {0.f,0.f,0.f,0.f,0.f,0.f,0.f,0.f};
  v8f o[4];
  #pragma unroll
  for (int t = 0; t < 4; ++t) o[t] = z;
  float mrow[8], lrow[8];
  #pragma unroll
  for (int j = 0; j < 8; ++j) { mrow[j] = -1e30f; lrow[j] = 0.f; }

  const int col = lane & 15;
  const int rb  = (lane >> 4) * 8;
  const float inv_scale = 1.0f / SCALE_F;
  const int kend = q0 + 16;                 // causal: keys <= q0+15
  const unsigned ldsV = lds_addr_of(&Vl[wv][lane * VDIM]);

  for (int kb = 0; kb < kend; kb += 32) {
    // previous iteration's V gathers (ds_loads) must land before async rewrites Vl
    wait_ds0();
    // async-stage V block: lane -> one key row of 64 bf16 (128B); overlaps QK+softmax
    async_copy64B(ldsV,      Vc + (size_t)(kb + lane) * VDIM);
    async_copy64B(ldsV + 64, Vc + (size_t)(kb + lane) * VDIM + 32);
    // prefetch next key block's K rows into cache
    if (kb + 32 < kend)
      __builtin_prefetch(Kc + (size_t)(kb + 32 + lane) * QKD, 0, 1);
    // scores S[16q x 32k] = Q @ K^T (two 16-wide n-tiles, 3 k-steps)
    v8f s0 = z, s1 = z;
    {
      const int hi = (lane >> 4) * 16;
      const u16* k0r = Kc + (size_t)(kb + col)      * QKD + hi;
      const u16* k1r = Kc + (size_t)(kb + 16 + col) * QKD + hi;
      #pragma unroll
      for (int ks = 0; ks < 3; ++ks) {
        v16bf b0 = cat88(*(const v8bf*)(k0r + ks * 32), *(const v8bf*)(k0r + ks * 32 + 8));
        v16bf b1 = cat88(*(const v8bf*)(k1r + ks * 32), *(const v8bf*)(k1r + ks * 32 + 8));
        s0 = wmma_bf16(qf[ks], b0, s0);
        s1 = wmma_bf16(qf[ks], b1, s1);
      }
    }
    // causal mask + scale (matches reference: mask -> -1e9, then /SCALE)
    #pragma unroll
    for (int j = 0; j < 8; ++j) {
      const int qrow = q0 + rb + j;
      s0[j] = (kb + col      > qrow) ? (-1.0e9f * inv_scale) : s0[j] * inv_scale;
      s1[j] = (kb + 16 + col > qrow) ? (-1.0e9f * inv_scale) : s1[j] * inv_scale;
    }
    // online softmax: row reductions across 16-lane halves (wave32)
    float rm[8];
    #pragma unroll
    for (int j = 0; j < 8; ++j) rm[j] = fmaxf(s0[j], s1[j]);
    #pragma unroll
    for (int off = 1; off < 16; off <<= 1)
      #pragma unroll
      for (int j = 0; j < 8; ++j) rm[j] = fmaxf(rm[j], __shfl_xor(rm[j], off, 32));
    float sc[8];
    #pragma unroll
    for (int j = 0; j < 8; ++j) {
      float mn = fmaxf(mrow[j], rm[j]);
      sc[j] = __expf(mrow[j] - mn);
      mrow[j] = mn;
      s0[j] = __expf(s0[j] - mn);
      s1[j] = __expf(s1[j] - mn);
    }
    float rs[8];
    #pragma unroll
    for (int j = 0; j < 8; ++j) rs[j] = s0[j] + s1[j];
    #pragma unroll
    for (int off = 1; off < 16; off <<= 1)
      #pragma unroll
      for (int j = 0; j < 8; ++j) rs[j] += __shfl_xor(rs[j], off, 32);
    #pragma unroll
    for (int j = 0; j < 8; ++j) lrow[j] = lrow[j] * sc[j] + rs[j];
    #pragma unroll
    for (int t = 0; t < 4; ++t)
      #pragma unroll
      for (int j = 0; j < 8; ++j) o[t][j] *= sc[j];
    // P (C-layout) -> LDS bf16, re-read as A-fragment
    {
      u16* P = Pl[wv];
      #pragma unroll
      for (int j = 0; j < 8; ++j) {
        const int r = rb + j;
        P[r * 32 + col]      = f2bf(s0[j]);
        P[r * 32 + 16 + col] = f2bf(s1[j]);
      }
    }
    // P stores + async V copy must both be visible before the gathers below
    asm volatile("s_wait_dscnt 0x0\n\ts_wait_asynccnt 0x0" ::: "memory");
    v16bf pf = frag_a_rowmajor(Pl[wv], 32);
    // O += P @ V  (contraction over 32 keys, 4 n-tiles of 16 v-dims)
    {
      const int kb16 = (lane >> 4) * 16;
      #pragma unroll
      for (int t = 0; t < 4; ++t) {
        u16x16 u;
        #pragma unroll
        for (int i = 0; i < 16; ++i) u[i] = Vl[wv][(kb16 + i) * VDIM + t * 16 + col];
        o[t] = wmma_bf16(pf, __builtin_bit_cast(v16bf, u), o[t]);
      }
    }
  }
  // epilogue: O[b, s, h*64 + vd]  (layout [NROW, NH*VDIM] for final GEMM)
  #pragma unroll
  for (int t = 0; t < 4; ++t)
    #pragma unroll
    for (int j = 0; j < 8; ++j) {
      const int gm = b * SEQ + q0 + rb + j;
      const int gc = hh * VDIM + t * 16 + col;
      Og[(size_t)gm * (NH * VDIM) + gc] = f2bf(o[t][j] / lrow[j]);
    }
}

// ---------- host launcher ----------
extern "C" void kernel_launch(void* const* d_in, const int* in_sizes, int n_in,
                              void* d_out, int out_size, void* d_ws, size_t ws_size,
                              hipStream_t stream)
{
  const float* h    = (const float*)d_in[0];
  // d_in[1] = mask (tril) -> causal handled analytically
  const float* Wdkv = (const float*)d_in[2];
  const float* bdkv = (const float*)d_in[3];
  const float* Wuk  = (const float*)d_in[4];
  const float* buk  = (const float*)d_in[5];
  const float* Wuv  = (const float*)d_in[6];
  const float* buv  = (const float*)d_in[7];
  const float* Wdq  = (const float*)d_in[8];
  const float* bdq  = (const float*)d_in[9];
  const float* Wuq  = (const float*)d_in[10];
  const float* buq  = (const float*)d_in[11];
  const float* Wqr  = (const float*)d_in[12];
  const float* bqr  = (const float*)d_in[13];
  const float* Wkr  = (const float*)d_in[14];
  const float* bkr  = (const float*)d_in[15];
  const float* Wfc  = (const float*)d_in[16];
  const float* bfc  = (const float*)d_in[17];
  float* out = (float*)d_out;

  char* p = (char*)d_ws;
  auto carve = [&](size_t elems) -> u16* {
    u16* r = (u16*)p;
    p += (elems * sizeof(u16) + 255) & ~(size_t)255;
    return r;
  };
  u16* h_bf    = carve((size_t)NROW * DIM);
  u16* Wdkv_bf = carve((size_t)DDOWN * DIM);
  u16* Wdq_bf  = carve((size_t)DDOWN * DIM);
  u16* Wuk_bf  = carve((size_t)DUP * DDOWN);
  u16* Wuv_bf  = carve((size_t)DUP * DDOWN);
  u16* Wuq_bf  = carve((size_t)DUP * DDOWN);
  u16* Wqr_bf  = carve((size_t)(RDIM * NH) * DDOWN);
  u16* Wkr_bf  = carve((size_t)RDIM * DIM);
  u16* Wfc_bf  = carve((size_t)DIM * DUP);
  u16* ckv     = carve((size_t)NROW * DDOWN);
  u16* cq      = carve((size_t)NROW * DDOWN);
  u16* kc_flat = carve((size_t)NROW * DUP);
  u16* vc_flat = carve((size_t)NROW * DUP);
  u16* qc_flat = carve((size_t)NROW * DUP);
  u16* qr_flat = carve((size_t)NROW * (RDIM * NH));
  u16* kr_flat = carve((size_t)NROW * RDIM);
  u16* Qbuf    = carve((size_t)BSZ * NH * SEQ * QKD);
  u16* Kbuf    = carve((size_t)BSZ * NH * SEQ * QKD);
  u16* Vbuf    = carve((size_t)BSZ * NH * SEQ * VDIM);
  u16* Obuf    = carve((size_t)NROW * (NH * VDIM));

  auto cvt = [&](const float* s, u16* d, int n) {
    cvt_f32_bf16_kernel<<<(n + 255) / 256, 256, 0, stream>>>(s, d, n);
  };
  cvt(h,    h_bf,    NROW * DIM);
  cvt(Wdkv, Wdkv_bf, DDOWN * DIM);
  cvt(Wdq,  Wdq_bf,  DDOWN * DIM);
  cvt(Wuk,  Wuk_bf,  DUP * DDOWN);
  cvt(Wuv,  Wuv_bf,  DUP * DDOWN);
  cvt(Wuq,  Wuq_bf,  DUP * DDOWN);
  cvt(Wqr,  Wqr_bf,  RDIM * NH * DDOWN);
  cvt(Wkr,  Wkr_bf,  RDIM * DIM);
  cvt(Wfc,  Wfc_bf,  DIM * DUP);

  auto gemm = [&](const u16* A, const u16* B, const float* bias,
                  u16* Cb, float* Cf, int M, int N, int K) {
    dim3 g((N + 127) / 128, M / 128);
    gemm_bf16_kernel<<<g, 256, 0, stream>>>(A, B, bias, Cb, Cf, M, N, K);
  };
  // down projections
  gemm(h_bf, Wdkv_bf, bdkv, ckv, nullptr, NROW, DDOWN, DIM);
  gemm(h_bf, Wdq_bf,  bdq,  cq,  nullptr, NROW, DDOWN, DIM);
  // up projections
  gemm(ckv, Wuk_bf, buk, kc_flat, nullptr, NROW, DUP, DDOWN);
  gemm(ckv, Wuv_bf, buv, vc_flat, nullptr, NROW, DUP, DDOWN);
  gemm(cq,  Wuq_bf, buq, qc_flat, nullptr, NROW, DUP, DDOWN);
  gemm(cq,  Wqr_bf, bqr, qr_flat, nullptr, NROW, RDIM * NH, DDOWN);
  gemm(h_bf, Wkr_bf, bkr, kr_flat, nullptr, NROW, RDIM, DIM);

  // head layout + rope
  {
    int n = NROW * NH * CDIM;
    scatter_heads_kernel<<<(n + 255) / 256, 256, 0, stream>>>(kc_flat, Kbuf, CDIM, QKD);
    scatter_heads_kernel<<<(n + 255) / 256, 256, 0, stream>>>(vc_flat, Vbuf, VDIM, VDIM);
    scatter_heads_kernel<<<(n + 255) / 256, 256, 0, stream>>>(qc_flat, Qbuf, CDIM, QKD);
  }
  rope_q_kernel<<<(NROW * NH * 16 + 255) / 256, 256, 0, stream>>>(qr_flat, Qbuf);
  rope_k_bcast_kernel<<<(NROW * 16 + 255) / 256, 256, 0, stream>>>(kr_flat, Kbuf);

  // flash attention: B*H*(S/16) 16-row tiles, 8 waves per block
  mla_attn_kernel<<<(BSZ * NH * (SEQ / 16)) / 8, 256, 0, stream>>>(Qbuf, Kbuf, Vbuf, Obuf);

  // final projection, fp32 output
  gemm(Obuf, Wfc_bf, bfc, nullptr, out, NROW, DIM, DUP);
}
